// EventFFTViT5_50019189129725
// MI455X (gfx1250) — compile-verified, matrix operand order checked by reading, not run on hardware
//
#include <hip/hip_runtime.h>
#include <hip/hip_bf16.h>

// ---------------------------------------------------------------------------
// FSAS_V5 fused forward for MI455X (gfx1250, wave32, WMMA + TDM).
// One workgroup (256 thr = 8 waves) owns one 8x8 patch of one batch image.
// All intermediates live in LDS; HBM traffic = read x + write out (~134 MB).
// Channel GEMMs: v_wmma_f32_16x16x32_f16 (f32 accumulate).
// Weight slabs staged into LDS by the Tensor Data Mover (tensor_load_to_lds,
// TENSORcnt), overlapped with VALU work.
// ---------------------------------------------------------------------------

typedef _Float16 half_t;
typedef __attribute__((ext_vector_type(16))) _Float16 v16h;
typedef __attribute__((ext_vector_type(8)))  _Float16 v8h;
typedef __attribute__((ext_vector_type(8)))  float    v8f;
typedef __attribute__((ext_vector_type(4)))  unsigned int u32x4;
typedef __attribute__((ext_vector_type(8)))  int      i32x8;
typedef __attribute__((ext_vector_type(4)))  int      i32x4;

#define TPB 256

// ---- LDS layout (dynamic shared), all regions 16B aligned ----
#define XS_OFF    0                        // f16 [112 px][64 ic]   (halo 10x10 padded to 112)
#define WBUF_OFF  (XS_OFF   + 112*64*2)    // f16 [128 oc][64 ic]   (union: wdw f32[128*9], w_proj f16[64][128])
#define HID_OFF   (WBUF_OFF + 16384)       // f16 [128 oc][128 px]  (union: corr f32[128][64], f32 weight staging)
#define QBUF_OFF  (HID_OFF  + 32768)       // f32 [128][64]         (union: sbuf f16 [64 px][128 c])
#define KBUF_OFF  (QBUF_OFF + 32768)       // f32 [128][64]         (union: f32 w_proj staging)
#define VBUF_OFF  (KBUF_OFF + 32768)       // f32 [128][64]
#define RED_OFF   (VBUF_OFF + 32768)       // f32 [64] per-pixel reductions
#define SMEM_BYTES (RED_OFF + 64*4)        // ~158 KB (WGP has 320 KB)

__device__ __forceinline__ v8f wmma_f16(v16h a, v16h b, v8f c) {
  // 8 args: (neg_a, A, neg_b, B, c_mod, C, reuse_a, reuse_b)
  return __builtin_amdgcn_wmma_f32_16x16x32_f16(false, a, false, b, (short)0, c,
                                                false, false);
}

// A/B fragment loader per ISA 7.12.2 (16-bit 16x32 A layout):
// lane<16 : row M=lane,    halves K = kbase+{0..7, 16..23}
// lane>=16: row M=lane-16, halves K = kbase+{8..15,24..31}
// B operand uses the mirrored layout with rows = N (we store B pixel-major).
__device__ __forceinline__ v16h load_frag(const half_t* base, int row0,
                                          int rowstride, int kbase, int lane) {
  int m    = lane & 15;
  int kofs = kbase + ((lane & 16) ? 8 : 0);
  const half_t* p = base + (row0 + m) * rowstride + kofs;
  v8h lo = *(const v8h*)(p);        // ds_load_b128
  v8h hi = *(const v8h*)(p + 16);   // ds_load_b128
  v16h r;
#pragma unroll
  for (int i = 0; i < 8; ++i) { r[i] = lo[i]; r[i + 8] = hi[i]; }
  return r;
}

// 1-D Tensor Data Mover copy: nelem f32 from global -> LDS (D# per ISA ch.8).
// Issued once per calling wave; tracked by TENSORcnt. TDM ignores EXEC.
__device__ __forceinline__ void tdm_load_1d(const void* gptr, unsigned lds_off,
                                            unsigned nelem, unsigned tensor_elems) {
  unsigned long long ga = (unsigned long long)gptr;
  u32x4 g0;
  g0[0] = 1u;                                         // count=1 (valid), user mode
  g0[1] = lds_off;                                    // lds_addr (bytes)
  g0[2] = (unsigned)(ga & 0xFFFFFFFFu);               // global_addr[31:0]
  g0[3] = (unsigned)((ga >> 32) & 0x1FFFFFFu)         // global_addr[56:32]
        | (2u << 30);                                 // type=2 ("image")
  i32x8 g1;
  g1[0] = (int)(2u << 16);                            // wg_mask=0, data_size=2 (4B)
  g1[1] = (int)((tensor_elems & 0xFFFFu) << 16);      // tensor_dim0[15:0] @bits63:48
  g1[2] = (int)((tensor_elems >> 16) & 0xFFFFu);      // tensor_dim0[31:16]; tensor_dim1=0
  g1[3] = (int)((nelem & 0xFFFFu) << 16);             // tile_dim0 @bits127:112
  g1[4] = 0;                                          // tile_dim1/2 = 0 (1-D tile)
  g1[5] = 0; g1[6] = 0; g1[7] = 0;                    // strides unused (1-D)
  i32x4 z = {0, 0, 0, 0};
#if defined(__clang_major__) && (__clang_major__ >= 23)
  i32x8 z8 = {0, 0, 0, 0, 0, 0, 0, 0};
  __builtin_amdgcn_tensor_load_to_lds(g0, g1, z, z, z8, 0);
#else
  __builtin_amdgcn_tensor_load_to_lds(g0, g1, z, z, 0);
#endif
}

__global__ void __launch_bounds__(TPB)
fsas_fused(const float* __restrict__ x, const float* __restrict__ w_hidden,
           const float* __restrict__ w_dw, const float* __restrict__ w_proj,
           const float* __restrict__ g_norm, const float* __restrict__ g_qnorm,
           const float* __restrict__ g_knorm, float* __restrict__ out) {
  extern __shared__ char smem[];
  half_t* xs   = (half_t*)(smem + XS_OFF);
  half_t* wbuf = (half_t*)(smem + WBUF_OFF);
  float*  wdw  = (float*)(smem + WBUF_OFF);
  half_t* hid  = (half_t*)(smem + HID_OFF);
  float*  corr = (float*)(smem + HID_OFF);
  float*  wstg = (float*)(smem + HID_OFF);   // f32 weight staging (TDM dest)
  float*  qbuf = (float*)(smem + QBUF_OFF);
  half_t* sbuf = (half_t*)(smem + QBUF_OFF);
  float*  kbuf = (float*)(smem + KBUF_OFF);
  float*  pstg = (float*)(smem + KBUF_OFF);  // f32 w_proj staging (TDM dest)
  float*  vbuf = (float*)(smem + VBUF_OFF);
  float*  red  = (float*)(smem + RED_OFF);

  const int tid  = threadIdx.x;
  const int lane = tid & 31;
  const int wv   = tid >> 5;          // 8 waves
  const int pid  = blockIdx.x;        // 0..1023 patch id
  const int b    = blockIdx.y;        // 0..3
  const int py   = pid >> 5, pxc = pid & 31;
  const int gy0  = py * 8, gx0 = pxc * 8;

  // ---- Phase 0: TDM-stage group-0 weights while loading the x halo tile ----
  if (wv == 0)
    tdm_load_1d(w_hidden, (unsigned)(uintptr_t)wstg, 128 * 64, 384 * 64);

  for (int idx = tid; idx < 112 * 64; idx += TPB) {
    int hp = idx >> 6, ic = idx & 63;
    float v = 0.f;
    if (hp < 100) {
      int hy = hp / 10, hx = hp % 10;
      int gy = gy0 - 1 + hy, gx = gx0 - 1 + hx;
      if (gy >= 0 && gy < 256 && gx >= 0 && gx < 256)
        v = x[(((size_t)b * 64 + ic) * 256 + gy) * 256 + gx];
    }
    xs[hp * 64 + ic] = (half_t)v;
  }
  __builtin_prefetch(w_dw + tid, 0, 1);    // dw weights are L2-resident

  // ---- q/k/v channel groups (128 out-channels each) ----
  for (int g = 0; g < 3; ++g) {
    if (wv == 0) {
      if (g > 0)   // stage this group's expand-weight slab via TDM
        tdm_load_1d(w_hidden + (size_t)g * 128 * 64,
                    (unsigned)(uintptr_t)wstg, 128 * 64, 384 * 64);
      __builtin_amdgcn_s_wait_tensorcnt(0);
    }
    __syncthreads();

    // convert staged f32 slab -> f16 operand buffer
    for (int idx = tid; idx < 128 * 64; idx += TPB)
      wbuf[idx] = (half_t)wstg[idx];
    __syncthreads();

    // GEMM1: hid[128 oc][112 px] = wbuf[128][64] x xs  (wave wv -> M-tile wv)
    {
      const int m0 = wv * 16;
      for (int nt = 0; nt < 7; ++nt) {
        const int n0 = nt * 16;
        v8f acc = {};
#pragma unroll
        for (int ks = 0; ks < 2; ++ks) {
          v16h a  = load_frag(wbuf, m0, 64, ks * 32, lane);
          v16h bf = load_frag(xs,  n0, 64, ks * 32, lane);
          acc = wmma_f16(a, bf, acc);
        }
        // C/D layout: VGPR r -> M = r + (lane>=16 ? 8 : 0), N = lane&15
        int n  = n0 + (lane & 15);
        int mb = m0 + ((lane & 16) ? 8 : 0);
#pragma unroll
        for (int r = 0; r < 8; ++r)
          hid[(mb + r) * 128 + n] = (half_t)acc[r];
      }
    }
    __syncthreads();

    // depthwise weights (reuse wbuf region as f32); zero per-pixel reduction
    for (int idx = tid; idx < 128 * 9; idx += TPB)
      wdw[idx] = w_dw[(size_t)(g * 128) * 9 + idx];
    if (tid < 64) red[tid] = 0.f;
    __syncthreads();

    float* tgt = (g == 0) ? qbuf : (g == 1) ? kbuf : vbuf;

    // depthwise 3x3 on inner 8x8 from halo; accumulate sum-of-squares (q,k)
    for (int idx = tid; idx < 128 * 64; idx += TPB) {
      int oc = idx >> 6, ip = idx & 63;
      int hy = (ip >> 3) + 1, hx = (ip & 7) + 1;
      const half_t* hrow = hid + oc * 128;
      const float*  wr   = wdw + oc * 9;
      float acc = 0.f;
#pragma unroll
      for (int dy = -1; dy <= 1; ++dy)
#pragma unroll
        for (int dx = -1; dx <= 1; ++dx)
          acc += wr[(dy + 1) * 3 + (dx + 1)] *
                 (float)hrow[(hy + dy) * 10 + (hx + dx)];
      tgt[oc * 64 + ip] = acc;
      if (g < 2) atomicAdd(&red[ip], acc * acc);   // ds_add_f32
    }
    __syncthreads();

    if (g < 2) {
      // RMSNorm over 128 channels + 2D RoPE (adjacent channel pairs)
      const float* gs = (g == 0) ? g_qnorm : g_knorm;
      for (int idx = tid; idx < 64 * 64; idx += TPB) {
        int pr = idx >> 6, ip = idx & 63;
        int c0 = pr * 2, c1 = c0 + 1;
        float invr = rsqrtf(red[ip] * (1.f / 128.f) + 1e-6f);
        float x0 = tgt[c0 * 64 + ip] * invr * gs[c0];
        float x1 = tgt[c1 * 64 + ip] * invr * gs[c1];
        int gy = gy0 + (ip >> 3), gx = gx0 + (ip & 7);
        float pos = (c0 < 64) ? (float)gy : (float)gx;
        int   j   = (c0 & 63) >> 1;
        float ang = pos * __expf(-(float)j * 0.28782313662425572f); // ln(1e4)/32
        float sn, cs;
        __sincosf(ang, &sn, &cs);
        tgt[c0 * 64 + ip] = x0 * cs - x1 * sn;     // rotate_half
        tgt[c1 * 64 + ip] = x1 * cs + x0 * sn;
      }
    }
    __syncthreads();
  }

  // ---- 8x8 circular convolution per channel: corr = irfft2(rfft2(q)*rfft2(k))
  if (tid < 64) red[tid] = 0.f;
  __syncthreads();
  {
    int c = tid >> 1;                  // channel 0..127
    int mbase = (tid & 1) * 4;         // half the output rows each
    const float* qr = qbuf + c * 64;
    const float* kr = kbuf + c * 64;
    for (int mm = 0; mm < 4; ++mm) {
      int m = mbase + mm;
      for (int n = 0; n < 8; ++n) {
        float acc = 0.f;
#pragma unroll
        for (int i = 0; i < 8; ++i) {
          int r = ((m - i) & 7) * 8;
#pragma unroll
          for (int j = 0; j < 8; ++j)
            acc += qr[i * 8 + j] * kr[r + ((n - j) & 7)];
        }
        corr[c * 64 + m * 8 + n] = acc;       // overwrites hid region (free)
        atomicAdd(&red[m * 8 + n], acc * acc);
      }
    }
  }
  __syncthreads();

  // ---- TDM-stage w_proj (into free kbuf region) overlapped with epilogue ----
  if (wv == 0)
    tdm_load_1d(w_proj, (unsigned)(uintptr_t)pstg, 64 * 128, 64 * 128);

  // s = RMSNorm(corr)*g_norm * v  -> f16 pixel-major (overwrites qbuf region)
  for (int idx = tid; idx < 128 * 64; idx += TPB) {
    int c = idx >> 6, ip = idx & 63;
    float invr = rsqrtf(red[ip] * (1.f / 128.f) + 1e-6f);
    float s = corr[c * 64 + ip] * invr * g_norm[c] * vbuf[c * 64 + ip];
    sbuf[ip * 128 + c] = (half_t)s;
  }
  if (wv == 0) __builtin_amdgcn_s_wait_tensorcnt(0);
  __syncthreads();

  for (int idx = tid; idx < 64 * 128; idx += TPB)    // f32 -> f16 operand
    wbuf[idx] = (half_t)pstg[idx];
  __syncthreads();

  // ---- GEMM2: out[64 oc][64 px] = w_proj[64][128] x s[128][64] ----
  for (int rep = 0; rep < 2; ++rep) {
    int t  = wv * 2 + rep;                    // 16 tiles over 8 waves
    int m0 = (t >> 2) * 16, n0 = (t & 3) * 16;
    v8f acc = {};
#pragma unroll
    for (int ks = 0; ks < 4; ++ks) {
      v16h a  = load_frag(wbuf, m0, 128, ks * 32, lane);
      v16h bf = load_frag(sbuf, n0, 128, ks * 32, lane);
      acc = wmma_f16(a, bf, acc);
    }
    int ip = n0 + (lane & 15);
    int gy = gy0 + (ip >> 3), gx = gx0 + (ip & 7);
    int mb = m0 + ((lane & 16) ? 8 : 0);
#pragma unroll
    for (int r = 0; r < 8; ++r)
      out[(((size_t)b * 64 + (mb + r)) * 256 + gy) * 256 + gx] = acc[r];
  }
}

extern "C" void kernel_launch(void* const* d_in, const int* in_sizes, int n_in,
                              void* d_out, int out_size, void* d_ws,
                              size_t ws_size, hipStream_t stream) {
  (void)in_sizes; (void)n_in; (void)out_size; (void)d_ws; (void)ws_size;
  const float* x        = (const float*)d_in[0];
  const float* w_hidden = (const float*)d_in[1];
  const float* w_dw     = (const float*)d_in[2];
  const float* w_proj   = (const float*)d_in[3];
  const float* g_norm   = (const float*)d_in[4];
  const float* g_qnorm  = (const float*)d_in[5];
  const float* g_knorm  = (const float*)d_in[6];
  float* outp = (float*)d_out;

  // allow >64KB dynamic LDS (WGP has 320KB)
  hipFuncSetAttribute((const void*)fsas_fused,
                      hipFuncAttributeMaxDynamicSharedMemorySize, SMEM_BYTES);

  dim3 grid(32 * 32, 4);   // one block per (patch, batch)
  fsas_fused<<<grid, TPB, SMEM_BYTES, stream>>>(x, w_hidden, w_dw, w_proj,
                                                g_norm, g_qnorm, g_knorm, outp);
}
